// Gnn_68393059222279
// MI455X (gfx1250) — compile-verified, matrix operand order checked by reading, not run on hardware
//
#include <hip/hip_runtime.h>
#include <hip/hip_bf16.h>
#include <stdint.h>

// ---------------- problem constants ----------------
#define DIN   2304     // 3*D input feature dim
#define DD    768      // embed dim
#define NCLS  32
#define BN_EPS 1e-5f
#define SLOPE  0.2f

typedef unsigned short u16t;
typedef __attribute__((ext_vector_type(16))) __bf16 v16bf;
typedef __attribute__((ext_vector_type(8)))  float  v8f;
typedef unsigned int __attribute__((ext_vector_type(4))) u32x4;
typedef int          __attribute__((ext_vector_type(4))) i32x4;
typedef int          __attribute__((ext_vector_type(8))) i32x8;

union Frag16 { uint4 q[2]; v16bf v; };

// LDS B-tile row stride: 32 bf16 (64B) + 16B TDM pad = 80B = 40 elements.
#define BROW 40

// ---------------- helpers ----------------
__device__ __forceinline__ unsigned bf16_rne(float f) {
  unsigned u = __float_as_uint(f);
  return (u + 0x7FFFu + ((u >> 16) & 1u)) >> 16;
}
__device__ __forceinline__ void split_bf16(float x, u16t* hi, u16t* lo) {
  unsigned h = bf16_rne(x);
  float fh = __uint_as_float(h << 16);
  unsigned l = bf16_rne(x - fh);
  *hi = (u16t)h; *lo = (u16t)l;
}
// order-preserving float<->uint encoding for atomicMax-based segment max
__device__ __forceinline__ unsigned fenc(float f) {
  unsigned u = __float_as_uint(f);
  return (u & 0x80000000u) ? ~u : (u | 0x80000000u);
}
__device__ __forceinline__ float fdec(unsigned e) {
  unsigned u = (e & 0x80000000u) ? (e ^ 0x80000000u) : ~e;
  return __uint_as_float(u);
}
#define ENC_NEG_INF 0x007FFFFFu   // fenc(-inf)

// Fragment loader (global, row-major [rows x ld] bf16-as-u16). CDNA5 16-bit A
// layout: lanes 0-15 hold K 0..7 & 16..23, lanes 16-31 hold K 8..15 & 24..31.
__device__ __forceinline__ v16bf load_frag(const u16t* __restrict__ base, int row0,
                                           int ld, int kc, int lane, int maxRow) {
  int m    = lane & 15;
  int ksel = lane >> 4;
  int row  = row0 + m; row = row > maxRow ? maxRow : row;
  const u16t* p = base + (size_t)row * ld + kc + ksel * 8;
  Frag16 f;
  f.q[0] = *(const uint4*)(p);        // K +0..7
  f.q[1] = *(const uint4*)(p + 16);   // K +16..23
  return f.v;
}

// Fragment loader from the TDM-staged LDS tile (padded 80B rows; k-chunk local)
__device__ __forceinline__ v16bf lds_frag(const u16t* sm, int row0, int lane) {
  int r    = row0 + (lane & 15);
  int ksel = lane >> 4;
  const u16t* p = sm + r * BROW + ksel * 8;
  Frag16 f;
  f.q[0] = *(const uint4*)(p);
  f.q[1] = *(const uint4*)(p + 16);
  return f.v;
}

__device__ __forceinline__ v8f wmma_bf16(v16bf a, v16bf b, v8f c) {
  return __builtin_amdgcn_wmma_f32_16x16x32_bf16(false, a, false, b, (short)0, c,
                                                 false, false);
}

// ---------------- Tensor Data Mover: stage a 128x32 bf16 tile to LDS --------
__device__ __forceinline__ void tdm_issue(u32x4 g0, i32x8 g1) {
  i32x4 z4 = {};
#if defined(__clang_major__) && (__clang_major__ >= 23)
  i32x8 z8 = {};
  __builtin_amdgcn_tensor_load_to_lds(g0, g1, z4, z4, z8, 0);
#else
  __builtin_amdgcn_tensor_load_to_lds(g0, g1, z4, z4, 0);
#endif
}

// D# per ISA 8.3/8.4: 2D tensor, data_size=2B, tile 32x128, row stride ldb,
// LDS padding 4 DWORDs every 16 DWORDs (64B row -> 80B row, bank-conflict-free).
__device__ __forceinline__ void tdm_load_tile(unsigned lds_byte_addr,
                                              const u16t* gsrc, int ldb_elems) {
  unsigned long long ga = (unsigned long long)(uintptr_t)gsrc;
  u32x4 g0 = {};
  g0[0] = 1u;                                              // count=1, user mode
  g0[1] = lds_byte_addr;                                   // LDS dest (bytes)
  g0[2] = (unsigned)ga;                                    // global_addr[31:0]
  g0[3] = ((unsigned)(ga >> 32) & 0x01FFFFFFu) | 0x80000000u; // [56:32] | type=2
  i32x8 g1 = {};
  // data_size=1(2B)<<16 | pad_enable<<20 | pad_interval=3(16dw)<<22 | pad_amount=3(4dw)<<25
  g1[0] = (int)0x06D10000u;
  g1[1] = (int)(32u  << 16);   // tensor_dim0 = 32 (k-chunk)
  g1[2] = (int)(128u << 16);   // tensor_dim1 = 128 rows
  g1[3] = (int)(32u  << 16);   // tile_dim0 = 32
  g1[4] = (int)128;            // tile_dim1 = 128
  g1[5] = ldb_elems;           // tensor_dim0_stride (elements)
  tdm_issue(g0, g1);
}

__device__ __forceinline__ unsigned lds_offset_of(const void* p) {
  // flat->LDS mapping keeps the byte offset in addr[31:0] (ISA 10.2 aperture table)
  return (unsigned)(uintptr_t)p;
}

// ---------------- BatchNorm ----------------
__global__ __launch_bounds__(256)
void bn_stats_k(const float* __restrict__ x, int N, float* __restrict__ mu,
                float* __restrict__ rstd) {
  __shared__ float sS[256], sQ[256];
  int col = blockIdx.x * 64 + (threadIdx.x & 63);
  int rg  = threadIdx.x >> 6;
  float s = 0.f, q = 0.f;
  for (int r = rg; r < N; r += 4) {
    float v = x[(size_t)r * DIN + col];
    s += v; q += v * v;
  }
  sS[threadIdx.x] = s; sQ[threadIdx.x] = q;
  __syncthreads();
  if (threadIdx.x < 64) {
    int c = blockIdx.x * 64 + threadIdx.x;
    float ts = 0.f, tq = 0.f;
    for (int g = 0; g < 4; ++g) { ts += sS[g * 64 + threadIdx.x]; tq += sQ[g * 64 + threadIdx.x]; }
    float m   = ts / (float)N;
    float var = tq / (float)N - m * m;   // biased variance (jnp.var)
    mu[c]   = m;
    rstd[c] = rsqrtf(var + BN_EPS);
  }
}

__global__ __launch_bounds__(256)
void bn_norm_split_k(const float* __restrict__ x, const float* __restrict__ mu,
                     const float* __restrict__ rstd, const float* __restrict__ g,
                     const float* __restrict__ b, size_t total,
                     u16t* __restrict__ hi, u16t* __restrict__ lo) {
  size_t idx = (size_t)blockIdx.x * blockDim.x + threadIdx.x;
  if (idx >= total) return;
  int col = (int)(idx % DIN);
  float v = (x[idx] - mu[col]) * rstd[col] * g[col] + b[col];
  split_bf16(v, &hi[idx], &lo[idx]);
}

// transpose W [K x Nc] -> split bf16 Wt [Nc x K]
__global__ __launch_bounds__(256)
void wt_split_T_k(const float* __restrict__ W, int K, int Nc,
                  u16t* __restrict__ Thi, u16t* __restrict__ Tlo) {
  size_t idx = (size_t)blockIdx.x * blockDim.x + threadIdx.x;
  if (idx >= (size_t)K * Nc) return;
  int k = (int)(idx / Nc), n = (int)(idx % Nc);
  split_bf16(W[idx], &Thi[(size_t)n * K + k], &Tlo[(size_t)n * K + k]);
}

// ---------------- WMMA GEMMs (split-bf16, fp32 accumulate) ----------------
// C[M x ldc(=768)] (+)= A[M x K] * Bt[cols x K]^T.
// 128x128 block, 8 waves (4m x 2n). B tiles staged to LDS by the TDM,
// double-buffered; A streamed with direct b128 loads.
__global__ __launch_bounds__(256)
void gemm_bf16s_128x128(const u16t* __restrict__ Ahi, const u16t* __restrict__ Alo,
                        int lda,
                        const u16t* __restrict__ Bhi, const u16t* __restrict__ Blo,
                        int ldb, float* __restrict__ C, int ldc, int M, int K,
                        int accumulate) {
  __shared__ u16t smB[2][2][128 * BROW];   // [buf][hi/lo], 80B rows -> 40KB total
  const int lane = threadIdx.x & 31;
  const int wid  = threadIdx.x >> 5;
  const int wm = wid >> 1, wn = wid & 1;
  const int mbase  = blockIdx.y * 128 + wm * 32;
  const int nblock = blockIdx.x * 128;
  const int nwave  = wn * 64;
  const u16t* Bh0 = Bhi + (size_t)nblock * ldb;
  const u16t* Bl0 = Blo + (size_t)nblock * ldb;

  // prologue: TDM-stage first k-chunk of the B tile (hi+lo)
  if (wid == 0) {
    tdm_load_tile(lds_offset_of(&smB[0][0][0]), Bh0, ldb);
    tdm_load_tile(lds_offset_of(&smB[0][1][0]), Bl0, ldb);
    __builtin_amdgcn_s_wait_tensorcnt(0);
  }
  __syncthreads();

  v8f c[2][4] = {};
  int buf = 0;
  for (int kc = 0; kc < K; kc += 32, buf ^= 1) {
    const bool more = (kc + 32) < K;
    if (wid == 0 && more) {   // overlap next tile DMA with this step's WMMAs
      tdm_load_tile(lds_offset_of(&smB[buf ^ 1][0][0]), Bh0 + (kc + 32), ldb);
      tdm_load_tile(lds_offset_of(&smB[buf ^ 1][1][0]), Bl0 + (kc + 32), ldb);
    }
    v16bf ah[2], alo[2], bh[4], bl[4];
#pragma unroll
    for (int i = 0; i < 2; ++i) {
      ah[i]  = load_frag(Ahi, mbase + i * 16, lda, kc, lane, M - 1);
      alo[i] = load_frag(Alo, mbase + i * 16, lda, kc, lane, M - 1);
    }
#pragma unroll
    for (int j = 0; j < 4; ++j) {
      bh[j] = lds_frag(&smB[buf][0][0], nwave + j * 16, lane);
      bl[j] = lds_frag(&smB[buf][1][0], nwave + j * 16, lane);
    }
#pragma unroll
    for (int i = 0; i < 2; ++i)
#pragma unroll
      for (int j = 0; j < 4; ++j) {
        c[i][j] = wmma_bf16(ah[i],  bh[j], c[i][j]);   // hi*hi
        c[i][j] = wmma_bf16(ah[i],  bl[j], c[i][j]);   // hi*lo
        c[i][j] = wmma_bf16(alo[i], bh[j], c[i][j]);   // lo*hi
      }
    if (wid == 0 && more) __builtin_amdgcn_s_wait_tensorcnt(0);
    __syncthreads();   // release next buffer to all waves / protect reuse
  }

  const int rsel = lane >> 4, csel = lane & 15;
#pragma unroll
  for (int i = 0; i < 2; ++i)
#pragma unroll
    for (int j = 0; j < 4; ++j)
#pragma unroll
      for (int r = 0; r < 8; ++r) {
        int row = mbase + i * 16 + rsel * 8 + r;
        int col = nblock + nwave + j * 16 + csel;
        if (row < M) {
          size_t off = (size_t)row * ldc + col;
          C[off] = accumulate ? (C[off] + c[i][j][r]) : c[i][j][r];
        }
      }
}

// C[M x 32] = A[M x K] * Bt[32 x K]^T + bias ; 128 rows per block, 8 waves.
// (weights are L2-resident; direct global loads are fine here)
__global__ __launch_bounds__(256)
void gemm_bf16s_128x32(const u16t* __restrict__ Ahi, const u16t* __restrict__ Alo,
                       int lda,
                       const u16t* __restrict__ Bhi, const u16t* __restrict__ Blo,
                       int ldb, const float* __restrict__ bias,
                       float* __restrict__ C, int M, int K) {
  const int lane = threadIdx.x & 31;
  const int wid  = threadIdx.x >> 5;
  const int mbase = blockIdx.x * 128 + wid * 16;
  v8f c[2] = {};
  for (int kc = 0; kc < K; kc += 32) {
    v16bf ah  = load_frag(Ahi, mbase, lda, kc, lane, M - 1);
    v16bf alo = load_frag(Alo, mbase, lda, kc, lane, M - 1);
#pragma unroll
    for (int j = 0; j < 2; ++j) {
      v16bf bh = load_frag(Bhi, j * 16, ldb, kc, lane, 31);
      v16bf bl = load_frag(Blo, j * 16, ldb, kc, lane, 31);
      c[j] = wmma_bf16(ah,  bh, c[j]);
      c[j] = wmma_bf16(ah,  bl, c[j]);
      c[j] = wmma_bf16(alo, bh, c[j]);
    }
  }
  const int rsel = lane >> 4, csel = lane & 15;
#pragma unroll
  for (int j = 0; j < 2; ++j)
#pragma unroll
    for (int r = 0; r < 8; ++r) {
      int row = mbase + rsel * 8 + r;
      int col = j * 16 + csel;
      if (row < M) C[(size_t)row * NCLS + col] = c[j][r] + bias[col];
    }
}

// ---------------- attention scores ----------------
__global__ __launch_bounds__(256)
void rowdot_k(const float* __restrict__ X, const float* __restrict__ v1,
              const float* __restrict__ v2, float* __restrict__ o1,
              float* __restrict__ o2, int M) {
  int lane = threadIdx.x & 31, wid = threadIdx.x >> 5;
  int row = blockIdx.x * 8 + wid;
  if (row >= M) return;
  float s1 = 0.f, s2 = 0.f;
  for (int c = lane; c < DD; c += 32) {
    float x = X[(size_t)row * DD + c];
    s1 += x * v1[c]; s2 += x * v2[c];
  }
  for (int o = 16; o; o >>= 1) { s1 += __shfl_xor(s1, o); s2 += __shfl_xor(s2, o); }
  if (lane == 0) { o1[row] = s1; o2[row] = s2; }
}

// ---------------- edge (segment softmax + aggregate) ----------------
__device__ __forceinline__ void edge_sd(const int* __restrict__ ei, int E, int id,
                                        int* s, int* d) {
  if (id < E) { *s = ei[id]; *d = ei[E + id]; }
  else        { *s = *d = id - E; }           // appended self-loops
}

__global__ __launch_bounds__(256)
void init_seg_k(unsigned* __restrict__ menc, float* __restrict__ den, int N) {
  int i = blockIdx.x * blockDim.x + threadIdx.x;
  if (i < N) { menc[i] = ENC_NEG_INF; den[i] = 0.f; }
}
__global__ __launch_bounds__(256)
void zero_f32_k(float* __restrict__ p, size_t n) {
  size_t i = (size_t)blockIdx.x * blockDim.x + threadIdx.x;
  if (i < n) p[i] = 0.f;
}

__global__ __launch_bounds__(256)
void edge_score_max_k(const int* __restrict__ ei, int E, int T,
                      const float* __restrict__ al, const float* __restrict__ ar,
                      float* __restrict__ eb, unsigned* __restrict__ menc) {
  int id = blockIdx.x * blockDim.x + threadIdx.x;
  if (id >= T) return;
  int s, d; edge_sd(ei, E, id, &s, &d);
  float e = al[s] + ar[d];
  e = e > 0.f ? e : SLOPE * e;                 // leaky_relu
  eb[id] = e;
  atomicMax(&menc[d], fenc(e));
}

__global__ __launch_bounds__(256)
void edge_expsum_k(const int* __restrict__ ei, int E, int T,
                   const float* __restrict__ eb, const unsigned* __restrict__ menc,
                   float* __restrict__ den, float* __restrict__ exb) {
  int id = blockIdx.x * blockDim.x + threadIdx.x;
  if (id >= T) return;
  int s, d; edge_sd(ei, E, id, &s, &d);
  float ex = expf(eb[id] - fdec(menc[d]));
  exb[id] = ex;
  unsafeAtomicAdd(&den[d], ex);
}

// one block per edge; 768 columns strided over 256 threads
__global__ __launch_bounds__(256)
void edge_aggregate_k(const int* __restrict__ ei, int E, int T,
                      const float* __restrict__ exb, const float* __restrict__ den,
                      const float* __restrict__ xl, float* __restrict__ agg) {
  int id = blockIdx.x;
  int s, d; edge_sd(ei, E, id, &s, &d);
  float coef = exb[id] / (den[d] + 1e-16f);
  const float* xr = xl  + (size_t)s * DD;
  float*       o  = agg + (size_t)d * DD;
  for (int c = threadIdx.x; c < DD; c += 256)
    unsafeAtomicAdd(&o[c], coef * xr[c]);
}

// h = agg + bias -> bf16 split (and optional fp32 out for h2)
__global__ __launch_bounds__(256)
void split_bias_k(const float* __restrict__ agg, const float* __restrict__ bias,
                  size_t total, u16t* __restrict__ hi, u16t* __restrict__ lo,
                  float* __restrict__ outF) {
  size_t idx = (size_t)blockIdx.x * blockDim.x + threadIdx.x;
  if (idx >= total) return;
  int col = (int)(idx % DD);
  float v = agg[idx] + bias[col];
  if (outF) outF[idx] = v;
  split_bf16(v, &hi[idx], &lo[idx]);
}

// ---------------- masked CE + accuracy ----------------
__global__ __launch_bounds__(256)
void loss_acc_k(const float* __restrict__ pool, const float* __restrict__ dir,
                const int* __restrict__ tgt, int N, float* __restrict__ sc) {
  int r = blockIdx.x * blockDim.x + threadIdx.x;
  if (r >= N) return;
  int t = tgt[r];
  if (t < 0) return;
  const float* P = pool + (size_t)r * NCLS;
  const float* Q = dir  + (size_t)r * NCLS;
  float mp = P[0]; int bp = 0;
  float mq = Q[0]; int bq = 0;
  for (int k = 1; k < NCLS; ++k) {
    float vp = P[k]; if (vp > mp) { mp = vp; bp = k; }
    float vq = Q[k]; if (vq > mq) { mq = vq; bq = k; }
  }
  float sp = 0.f, sq = 0.f;
  for (int k = 0; k < NCLS; ++k) { sp += expf(P[k] - mp); sq += expf(Q[k] - mq); }
  float nllp = (mp + logf(sp)) - P[t];
  float nllq = (mq + logf(sq)) - Q[t];
  unsafeAtomicAdd(&sc[0], nllp);
  unsafeAtomicAdd(&sc[1], nllq);
  unsafeAtomicAdd(&sc[2], bp == t ? 1.f : 0.f);
  unsafeAtomicAdd(&sc[3], bq == t ? 1.f : 0.f);
  unsafeAtomicAdd(&sc[4], 1.f);
}

__global__ void finalize_k(const float* __restrict__ sc, float* __restrict__ out3) {
  if (threadIdx.x == 0 && blockIdx.x == 0) {
    float cnt = sc[4];
    out3[0] = sc[0] / cnt + sc[1] / cnt;  // loss = ce(pooler) + ce(direct)
    out3[1] = sc[2] / cnt;                // acc_gcn
    out3[2] = sc[3] / cnt;                // acc_lin
  }
}

// ---------------- launch ----------------
extern "C" void kernel_launch(void* const* d_in, const int* in_sizes, int n_in,
                              void* d_out, int out_size, void* d_ws, size_t ws_size,
                              hipStream_t stream) {
  const float* x      = (const float*)d_in[0];
  const int*   ei     = (const int*)  d_in[1];
  const int*   tgt    = (const int*)  d_in[2];
  const float* gamma  = (const float*)d_in[3];
  const float* beta   = (const float*)d_in[4];
  const float* W1     = (const float*)d_in[5];
  const float* a_src1 = (const float*)d_in[6];
  const float* a_dst1 = (const float*)d_in[7];
  const float* b1     = (const float*)d_in[8];
  const float* W2     = (const float*)d_in[9];
  const float* a_src2 = (const float*)d_in[10];
  const float* a_dst2 = (const float*)d_in[11];
  const float* b2     = (const float*)d_in[12];
  const float* poolW  = (const float*)d_in[13];
  const float* poolB  = (const float*)d_in[14];
  const float* dirW   = (const float*)d_in[15];
  const float* dirB   = (const float*)d_in[16];

  const int N = in_sizes[0] / DIN;
  const int E = in_sizes[1] / 2;
  const int T = E + N;                 // edges + self-loops

  // workspace carve (256B aligned)
  char* base = (char*)d_ws; size_t cur = 0;
  auto carve = [&](size_t bytes) -> void* {
    size_t o = (cur + 255) & ~(size_t)255; cur = o + bytes; return (void*)(base + o);
  };
  float* mu   = (float*)carve(DIN * 4);
  float* rstd = (float*)carve(DIN * 4);
  u16t* xnh = (u16t*)carve((size_t)N * DIN * 2);
  u16t* xnl = (u16t*)carve((size_t)N * DIN * 2);
  u16t* w1h = (u16t*)carve((size_t)DD * DIN * 2);      // [768 x 2304]
  u16t* w1l = (u16t*)carve((size_t)DD * DIN * 2);
  u16t* w2h = (u16t*)carve((size_t)DD * 4 * DD * 2);   // [768 x 3072]
  u16t* w2l = (u16t*)carve((size_t)DD * 4 * DD * 2);
  u16t* dwh = (u16t*)carve((size_t)NCLS * DIN * 2);    // [32 x 2304]
  u16t* dwl = (u16t*)carve((size_t)NCLS * DIN * 2);
  u16t* pwh = (u16t*)carve((size_t)NCLS * DD * 2);     // [32 x 768]
  u16t* pwl = (u16t*)carve((size_t)NCLS * DD * 2);
  float* xl  = (float*)carve((size_t)N * DD * 4);      // xl1 then xl2 (reused)
  float* al  = (float*)carve((size_t)N * 4);
  float* ar  = (float*)carve((size_t)N * 4);
  float* ebuf = (float*)carve((size_t)T * 4);
  float* exb  = (float*)carve((size_t)T * 4);
  unsigned* menc = (unsigned*)carve((size_t)N * 4);
  float* den  = (float*)carve((size_t)N * 4);
  float* agg  = (float*)carve((size_t)N * DD * 4);     // layer1 then layer2 (reused)
  u16t* h1h = (u16t*)carve((size_t)N * DD * 2);
  u16t* h1l = (u16t*)carve((size_t)N * DD * 2);
  u16t* h2h = (u16t*)carve((size_t)N * DD * 2);
  u16t* h2l = (u16t*)carve((size_t)N * DD * 2);
  float* sc = (float*)carve(8 * 4);

  float* out     = (float*)d_out;
  float* outH2   = out;
  float* outDir  = out + (size_t)N * DD;
  float* outPool = outDir + (size_t)N * NCLS;
  float* outScal = outPool + (size_t)N * NCLS;

  const size_t totX = (size_t)N * DIN;
  const size_t totH = (size_t)N * DD;
  const unsigned gX = (unsigned)((totX + 255) / 256);
  const unsigned gH = (unsigned)((totH + 255) / 256);
  const dim3 gGemm(DD / 128, (unsigned)((N + 127) / 128));
  const unsigned gRows32 = (unsigned)((N + 127) / 128);

  // 1) BatchNorm
  bn_stats_k<<<DIN / 64, 256, 0, stream>>>(x, N, mu, rstd);
  bn_norm_split_k<<<gX, 256, 0, stream>>>(x, mu, rstd, gamma, beta, totX, xnh, xnl);

  // 2) weight prep (transpose + hi/lo split)
  wt_split_T_k<<<(unsigned)(((size_t)DIN * DD + 255) / 256), 256, 0, stream>>>(W1, DIN, DD, w1h, w1l);
  wt_split_T_k<<<(unsigned)(((size_t)4 * DD * DD + 255) / 256), 256, 0, stream>>>(W2, 4 * DD, DD, w2h, w2l);
  wt_split_T_k<<<(unsigned)(((size_t)DIN * NCLS + 255) / 256), 256, 0, stream>>>(dirW, DIN, NCLS, dwh, dwl);
  wt_split_T_k<<<(unsigned)(((size_t)DD * NCLS + 255) / 256), 256, 0, stream>>>(poolW, DD, NCLS, pwh, pwl);

  // 3) GAT layer 1
  gemm_bf16s_128x128<<<gGemm, 256, 0, stream>>>(xnh, xnl, DIN, w1h, w1l, DIN, xl, DD, N, DIN, 0);
  rowdot_k<<<(unsigned)((N + 7) / 8), 256, 0, stream>>>(xl, a_src1, a_dst1, al, ar, N);
  init_seg_k<<<(unsigned)((N + 255) / 256), 256, 0, stream>>>(menc, den, N);
  zero_f32_k<<<gH, 256, 0, stream>>>(agg, totH);
  edge_score_max_k<<<(unsigned)((T + 255) / 256), 256, 0, stream>>>(ei, E, T, al, ar, ebuf, menc);
  edge_expsum_k<<<(unsigned)((T + 255) / 256), 256, 0, stream>>>(ei, E, T, ebuf, menc, den, exb);
  edge_aggregate_k<<<(unsigned)T, 256, 0, stream>>>(ei, E, T, exb, den, xl, agg);
  split_bias_k<<<gH, 256, 0, stream>>>(agg, b1, totH, h1h, h1l, (float*)nullptr);

  // 4) GAT layer 2: [h1 | xn] @ W2 as two K-range passes
  gemm_bf16s_128x128<<<gGemm, 256, 0, stream>>>(h1h, h1l, DD, w2h, w2l, 4 * DD, xl, DD, N, DD, 0);
  gemm_bf16s_128x128<<<gGemm, 256, 0, stream>>>(xnh, xnl, DIN, w2h + DD, w2l + DD, 4 * DD, xl, DD, N, DIN, 1);
  rowdot_k<<<(unsigned)((N + 7) / 8), 256, 0, stream>>>(xl, a_src2, a_dst2, al, ar, N);
  init_seg_k<<<(unsigned)((N + 255) / 256), 256, 0, stream>>>(menc, den, N);
  zero_f32_k<<<gH, 256, 0, stream>>>(agg, totH);
  edge_score_max_k<<<(unsigned)((T + 255) / 256), 256, 0, stream>>>(ei, E, T, al, ar, ebuf, menc);
  edge_expsum_k<<<(unsigned)((T + 255) / 256), 256, 0, stream>>>(ei, E, T, ebuf, menc, den, exb);
  edge_aggregate_k<<<(unsigned)T, 256, 0, stream>>>(ei, E, T, exb, den, xl, agg);
  split_bias_k<<<gH, 256, 0, stream>>>(agg, b2, totH, h2h, h2l, outH2);   // h2 -> output

  // 5) heads
  gemm_bf16s_128x32<<<gRows32, 256, 0, stream>>>(xnh, xnl, DIN, dwh, dwl, DIN, dirB, outDir, N, DIN);
  gemm_bf16s_128x32<<<gRows32, 256, 0, stream>>>(h2h, h2l, DD, pwh, pwl, DD, poolB, outPool, N, DD);

  // 6) loss + accuracy
  zero_f32_k<<<1, 32, 0, stream>>>(sc, 8);
  loss_acc_k<<<(unsigned)((N + 255) / 256), 256, 0, stream>>>(outPool, outDir, tgt, N, sc);
  finalize_k<<<1, 1, 0, stream>>>(sc, outScal);
}
